// TriOut_20194936225884
// MI455X (gfx1250) — matrix operand order repaired and checked by reading, not compiled, not run on hardware
//
#include <hip/hip_runtime.h>

// Triangle multiplicative update (outgoing), L=768, Z=128, C=32, all-bf16 WMMA.
constexpr int LSZ  = 768;
constexpr int ZDIM = 128;
constexpr int CDIM = 32;
constexpr int LLSZ = LSZ * LSZ;   // 589824 rows

typedef unsigned short u16;
typedef __attribute__((ext_vector_type(16))) __bf16       v16bf;
typedef __attribute__((ext_vector_type(8)))  float        v8f;
typedef __attribute__((ext_vector_type(4)))  unsigned int u32x4;
typedef __attribute__((ext_vector_type(4)))  float        f32x4;

union Frag  { v16bf v; u32x4 q[2]; u16 s[16]; };
union Pack8 { u32x4 q; u16 s[8]; };

__device__ __forceinline__ u16 f2bf(float f) {
  unsigned u = __builtin_bit_cast(unsigned, f);
  u += 0x7FFFu + ((u >> 16) & 1u);            // round-to-nearest-even
  return (u16)(u >> 16);
}
__device__ __forceinline__ float bf2f(u16 s) {
  return __builtin_bit_cast(float, (unsigned)s << 16);
}
__device__ __forceinline__ float sigm(float x) { return 1.0f / (1.0f + __expf(-x)); }
__device__ __forceinline__ v8f zero8() { v8f z = {0.f,0.f,0.f,0.f,0.f,0.f,0.f,0.f}; return z; }
__device__ __forceinline__ v8f wmma_bf(const Frag& a, const Frag& b, v8f c) {
  return __builtin_amdgcn_wmma_f32_16x16x32_bf16(false, a.v, false, b.v, (short)0, c,
                                                 false, false);
}

// ---------------- k0: f32 -> bf16 weight conversion --------------------------
__global__ void k0_cvt(const float* __restrict__ src, u16* __restrict__ dst, int n) {
  int i = blockIdx.x * blockDim.x + threadIdx.x;
  if (i < n) dst[i] = f2bf(src[i]);
}

// ---------------- k1: LayerNorm(z) + a/ag projections + sigmoid gate ---------
// 128 rows/block (8 waves, 16 rows each). zn tile in LDS (bf16, padded).
// a_t layout: [c][row] bf16 (channel-major -> both GEMM operands of k2).
__global__ __launch_bounds__(256) void k1_ln_proj(
    const float* __restrict__ z, const float* __restrict__ nw, const float* __restrict__ nb,
    const u16* __restrict__ a_wb, const float* __restrict__ a_b,
    const u16* __restrict__ ag_wb, const float* __restrict__ ag_b,
    u16* __restrict__ a_t)
{
  __shared__ u16 zn_s[128][136];
  const int t  = threadIdx.x;
  const int r0 = blockIdx.x << 7;

  { // LayerNorm over Z=128: two threads per row, 64 elements each
    const int row = t >> 1, half = t & 1;
    const float* zp = z + ((size_t)(r0 + row) << 7) + (half << 6);
    f32x4 v[16]; float s = 0.f, s2 = 0.f;
#pragma unroll
    for (int q = 0; q < 16; ++q) {
      v[q] = reinterpret_cast<const f32x4*>(zp)[q];
      s  += v[q].x + v[q].y + v[q].z + v[q].w;
      s2 += v[q].x*v[q].x + v[q].y*v[q].y + v[q].z*v[q].z + v[q].w*v[q].w;
    }
    s  += __shfl_xor(s, 1);
    s2 += __shfl_xor(s2, 1);
    const float mean = s * (1.f/128.f);
    const float inv  = rsqrtf(fmaxf(s2*(1.f/128.f) - mean*mean, 0.f) + 1e-5f);
#pragma unroll
    for (int q = 0; q < 16; ++q) {
      const int k = (half << 6) + (q << 2);
      zn_s[row][k+0] = f2bf((v[q].x - mean)*inv*nw[k+0] + nb[k+0]);
      zn_s[row][k+1] = f2bf((v[q].y - mean)*inv*nw[k+1] + nb[k+1]);
      zn_s[row][k+2] = f2bf((v[q].z - mean)*inv*nw[k+2] + nb[k+2]);
      zn_s[row][k+3] = f2bf((v[q].w - mean)*inv*nw[k+3] + nb[k+3]);
    }
  }
  __syncthreads();

  const int w = t >> 5, lane = t & 31;
  const int m0 = w << 4, lm = lane & 15, hi = lane >> 4;

  v8f acc_a[2] = { zero8(), zero8() };
  v8f acc_g[2] = { zero8(), zero8() };
#pragma unroll
  for (int ks = 0; ks < 4; ++ks) {
    const int kk = ks << 5;
    Frag fa;   // A 16x32 bf16: lane<16 -> K {0..15}; lane>=16 -> K {16..31}
    fa.q[0] = *reinterpret_cast<const u32x4*>(&zn_s[m0 + lm][kk + hi*16]);
    fa.q[1] = *reinterpret_cast<const u32x4*>(&zn_s[m0 + lm][kk + hi*16 + 8]);
#pragma unroll
    for (int nf = 0; nf < 2; ++nf) {
      const int n = (nf << 4) + lm;
      Frag fb;  // B 32x16: lane holds column n, 16 contiguous K from row-major W[n][k]
      const u16* bp = a_wb + n*ZDIM + kk + hi*16;
      fb.q[0] = *reinterpret_cast<const u32x4*>(bp);
      fb.q[1] = *reinterpret_cast<const u32x4*>(bp + 8);
      acc_a[nf] = wmma_bf(fa, fb, acc_a[nf]);
      const u16* gp = ag_wb + n*ZDIM + kk + hi*16;
      fb.q[0] = *reinterpret_cast<const u32x4*>(gp);
      fb.q[1] = *reinterpret_cast<const u32x4*>(gp + 8);
      acc_g[nf] = wmma_bf(fa, fb, acc_g[nf]);
    }
  }
#pragma unroll
  for (int nf = 0; nf < 2; ++nf) {
    const int n = (nf << 4) + lm;
    const float ab = a_b[n], gb = ag_b[n];
    Pack8 pk;
#pragma unroll
    for (int v = 0; v < 8; ++v)
      pk.s[v] = f2bf((acc_a[nf][v] + ab) * sigm(acc_g[nf][v] + gb));
    // D layout: m = v + 8*hi -> 8 consecutive rows -> one contiguous 16B store
    u16* dp = a_t + (size_t)n*LLSZ + r0 + m0 + hi*8;
    *reinterpret_cast<u32x4*>(dp) = pk.q;
  }
}

// ---------------- k2: O_c = A_c * A_c^T for all 32 channels ------------------
// a_t is 37.75 MB -> fully L2-resident; fragments loaded straight from L2.
// Block tile 128(i) x 256(j), 8 waves as 2x4; each wave a 64x64 output tile:
// acc[4][4] (128 VGPRs), A-frags reused over 4 N-tiles, B-frag over 4 M-tiles
// -> 1 global_load_b128 per WMMA (~32 FLOP per L2 byte).
__global__ __launch_bounds__(256) void k2_tri(const u16* __restrict__ a_t,
                                              u16* __restrict__ o_pre)
{
  const int c = blockIdx.z;
  const u16* A = a_t + (size_t)c * LLSZ;
  const int bi = blockIdx.x << 7;            // 128-row i tile
  const int bj = blockIdx.y << 8;            // 256-col j tile
  const int t = threadIdx.x, w = t >> 5, lane = t & 31;
  const int m0 = bi + ((w & 1) << 6);        // 2 waves along M (64 rows each)
  const int n0 = bj + ((w >> 1) << 6);       // 4 waves along N (64 cols each)
  const int lm = lane & 15, hi = lane >> 4;

  v8f acc[4][4];
#pragma unroll
  for (int mi = 0; mi < 4; ++mi)
#pragma unroll
    for (int ni = 0; ni < 4; ++ni) acc[mi][ni] = zero8();

  for (int kk = 0; kk < LSZ; kk += 32) {
    Frag fa[4];
#pragma unroll
    for (int mi = 0; mi < 4; ++mi) {
      const u16* ap = A + (size_t)(m0 + (mi << 4) + lm) * LSZ + kk + hi*16;
      fa[mi].q[0] = *reinterpret_cast<const u32x4*>(ap);
      fa[mi].q[1] = *reinterpret_cast<const u32x4*>(ap + 8);
    }
#pragma unroll
    for (int ni = 0; ni < 4; ++ni) {  // B = A^T: row n of A gives column n of B
      Frag fb;
      const u16* bp = A + (size_t)(n0 + (ni << 4) + lm) * LSZ + kk + hi*16;
      fb.q[0] = *reinterpret_cast<const u32x4*>(bp);
      fb.q[1] = *reinterpret_cast<const u32x4*>(bp + 8);
#pragma unroll
      for (int mi = 0; mi < 4; ++mi)
        acc[mi][ni] = wmma_bf(fa[mi], fb, acc[mi][ni]);
    }
  }
#pragma unroll
  for (int mi = 0; mi < 4; ++mi)
#pragma unroll
    for (int ni = 0; ni < 4; ++ni) {
      const int j = n0 + (ni << 4) + lm;
#pragma unroll
      for (int v = 0; v < 8; ++v) {
        const int i = m0 + (mi << 4) + hi*8 + v;
        o_pre[((size_t)i * LSZ + j) * CDIM + c] = f2bf(acc[mi][ni][v]);
      }
    }
}

// ---------------- k3: LN(o) @ o_w^T, gated by sigmoid(LN(z) @ g_w^T) ---------
// 128 rows/block. o-projection: K=32 (1 WMMA step); gate projection: K=128.
__global__ __launch_bounds__(256) void k3_out(
    const u16* __restrict__ o_pre,
    const float* __restrict__ z, const float* __restrict__ nw, const float* __restrict__ nb,
    const float* __restrict__ onw, const float* __restrict__ onb,
    const u16* __restrict__ o_wb, const float* __restrict__ o_b,
    const u16* __restrict__ g_wb, const float* __restrict__ g_b,
    float* __restrict__ out)
{
  __shared__ u16 zn_s[128][136];
  __shared__ u16 on_s[128][40];
  const int t  = threadIdx.x;
  const int r0 = blockIdx.x << 7;

  if (t < 128) { // LayerNorm over C=32 of o_pre (one thread per row)
    union { u32x4 q[4]; u16 s[32]; } buf;
    const u16* op = o_pre + ((size_t)(r0 + t) << 5);
#pragma unroll
    for (int qq = 0; qq < 4; ++qq) buf.q[qq] = reinterpret_cast<const u32x4*>(op)[qq];
    float s = 0.f, s2 = 0.f;
#pragma unroll
    for (int k = 0; k < 32; ++k) { const float f = bf2f(buf.s[k]); s += f; s2 += f*f; }
    const float mean = s * (1.f/32.f);
    const float inv  = rsqrtf(fmaxf(s2*(1.f/32.f) - mean*mean, 0.f) + 1e-5f);
#pragma unroll
    for (int k = 0; k < 32; ++k)
      on_s[t][k] = f2bf((bf2f(buf.s[k]) - mean)*inv*onw[k] + onb[k]);
  }
  { // re-derive zn for the gate (cheaper than a 151 MB gate buffer)
    const int row = t >> 1, half = t & 1;
    const float* zp = z + ((size_t)(r0 + row) << 7) + (half << 6);
    f32x4 v[16]; float s = 0.f, s2 = 0.f;
#pragma unroll
    for (int q = 0; q < 16; ++q) {
      v[q] = reinterpret_cast<const f32x4*>(zp)[q];
      s  += v[q].x + v[q].y + v[q].z + v[q].w;
      s2 += v[q].x*v[q].x + v[q].y*v[q].y + v[q].z*v[q].z + v[q].w*v[q].w;
    }
    s  += __shfl_xor(s, 1);
    s2 += __shfl_xor(s2, 1);
    const float mean = s * (1.f/128.f);
    const float inv  = rsqrtf(fmaxf(s2*(1.f/128.f) - mean*mean, 0.f) + 1e-5f);
#pragma unroll
    for (int q = 0; q < 16; ++q) {
      const int k = (half << 6) + (q << 2);
      zn_s[row][k+0] = f2bf((v[q].x - mean)*inv*nw[k+0] + nb[k+0]);
      zn_s[row][k+1] = f2bf((v[q].y - mean)*inv*nw[k+1] + nb[k+1]);
      zn_s[row][k+2] = f2bf((v[q].z - mean)*inv*nw[k+2] + nb[k+2]);
      zn_s[row][k+3] = f2bf((v[q].w - mean)*inv*nw[k+3] + nb[k+3]);
    }
  }
  __syncthreads();

  const int w = t >> 5, lane = t & 31;
  const int m0 = w << 4, lm = lane & 15, hi = lane >> 4;

  Frag fo;
  fo.q[0] = *reinterpret_cast<const u32x4*>(&on_s[m0 + lm][hi*16]);
  fo.q[1] = *reinterpret_cast<const u32x4*>(&on_s[m0 + lm][hi*16 + 8]);
  Frag fz[4];
#pragma unroll
  for (int ks = 0; ks < 4; ++ks) {
    fz[ks].q[0] = *reinterpret_cast<const u32x4*>(&zn_s[m0 + lm][(ks << 5) + hi*16]);
    fz[ks].q[1] = *reinterpret_cast<const u32x4*>(&zn_s[m0 + lm][(ks << 5) + hi*16 + 8]);
  }
#pragma unroll
  for (int nf = 0; nf < 8; ++nf) {
    const int n = (nf << 4) + lm;
    Frag fb;
    const u16* pb = o_wb + n*CDIM + hi*16;
    fb.q[0] = *reinterpret_cast<const u32x4*>(pb);
    fb.q[1] = *reinterpret_cast<const u32x4*>(pb + 8);
    v8f accO = wmma_bf(fo, fb, zero8());
    v8f accG = zero8();
#pragma unroll
    for (int ks = 0; ks < 4; ++ks) {
      const u16* pg = g_wb + (size_t)n*ZDIM + (ks << 5) + hi*16;
      fb.q[0] = *reinterpret_cast<const u32x4*>(pg);
      fb.q[1] = *reinterpret_cast<const u32x4*>(pg + 8);
      accG = wmma_bf(fz[ks], fb, accG);
    }
    const float ob = o_b[n], gb = g_b[n];
#pragma unroll
    for (int v = 0; v < 8; ++v) {
      const int r = r0 + m0 + hi*8 + v;
      out[((size_t)r << 7) + n] = (accO[v] + ob) * sigm(accG[v] + gb);
    }
  }
}

// ---------------------------------------------------------------------------
extern "C" void kernel_launch(void* const* d_in, const int* in_sizes, int n_in,
                              void* d_out, int out_size, void* d_ws, size_t ws_size,
                              hipStream_t stream) {
  (void)in_sizes; (void)n_in; (void)out_size; (void)ws_size;
  const float* z       = (const float*)d_in[0];
  const float* norm_w  = (const float*)d_in[1];
  const float* norm_b  = (const float*)d_in[2];
  const float* onorm_w = (const float*)d_in[3];
  const float* onorm_b = (const float*)d_in[4];
  const float* a_w     = (const float*)d_in[5];
  const float* a_b     = (const float*)d_in[6];
  const float* ag_w    = (const float*)d_in[7];
  const float* ag_b    = (const float*)d_in[8];
  const float* g_w     = (const float*)d_in[9];
  const float* g_b     = (const float*)d_in[10];
  const float* o_w     = (const float*)d_in[11];
  const float* o_b     = (const float*)d_in[12];
  float* out = (float*)d_out;

  // Workspace: bf16 weights (64 KB) | a_t [32][768*768] bf16 | o_pre [768*768][32] bf16
  char* ws = (char*)d_ws;
  u16* a_wb  = (u16*)(ws + 0);
  u16* ag_wb = (u16*)(ws + 8192);
  u16* g_wb  = (u16*)(ws + 16384);
  u16* o_wb  = (u16*)(ws + 49152);
  u16* a_t   = (u16*)(ws + 65536);
  u16* o_pre = (u16*)(ws + 65536 + (size_t)CDIM * LLSZ * 2);

  k0_cvt<<<16, 256, 0, stream>>>(a_w,  a_wb,  CDIM * ZDIM);
  k0_cvt<<<16, 256, 0, stream>>>(ag_w, ag_wb, CDIM * ZDIM);
  k0_cvt<<<64, 256, 0, stream>>>(g_w,  g_wb,  ZDIM * ZDIM);
  k0_cvt<<<16, 256, 0, stream>>>(o_w,  o_wb,  ZDIM * CDIM);

  k1_ln_proj<<<LLSZ / 128, 256, 0, stream>>>(z, norm_w, norm_b,
                                             a_wb, a_b, ag_wb, ag_b, a_t);
  k2_tri<<<dim3(LSZ / 128, LSZ / 256, CDIM), 256, 0, stream>>>(a_t, o_pre);
  k3_out<<<LLSZ / 128, 256, 0, stream>>>(o_pre, z, norm_w, norm_b, onorm_w, onorm_b,
                                         o_wb, o_b, g_wb, g_b, out);
}
